// MultiHeadAttention_23467701305650
// MI455X (gfx1250) — matrix operand order repaired and checked by reading, not compiled, not run on hardware
//
#include <hip/hip_runtime.h>
#include <hip/hip_bf16.h>

// Problem constants (from reference)
#define DIMV   1024
#define NHEAD  16
#define HDIM   64
#define SEQ    2048
#define BATCH  2
#define ROWS   (BATCH * SEQ)        // 4096 tokens
#define QKVO   (3 * DIMV)           // 3072
#define SCALEF 0.125f               // HDIM^-0.5

typedef __attribute__((ext_vector_type(16))) __bf16 v16bf;
typedef __attribute__((ext_vector_type(8)))  float  v8f;
typedef unsigned int v4u  __attribute__((ext_vector_type(4)));
typedef int          v8i_t __attribute__((ext_vector_type(8)));
typedef int          v4i_t __attribute__((ext_vector_type(4)));

#if defined(__has_builtin)
#if __has_builtin(__builtin_amdgcn_tensor_load_to_lds)
#define HAVE_TDM 1
#else
#define HAVE_TDM 0
#endif
#else
#define HAVE_TDM 0
#endif

union Frag {
    v16bf v;
    unsigned short s[16];
};

__device__ __forceinline__ unsigned short f2bf(float f) {
    unsigned int u = __float_as_uint(f);
    unsigned int r = u + 0x7fffu + ((u >> 16) & 1u);   // RNE
    return (unsigned short)(r >> 16);
}

#define WMMA_BF16(a, b, c) \
    __builtin_amdgcn_wmma_f32_16x16x32_bf16(false, (a), false, (b), (short)0, (c), false, false)

#if HAVE_TDM
// Issue one 2D TDM tile load: global (row-major, stride0 elements/row, 2-byte
// elements) -> LDS at lds_off.  Descriptor bitfields per CDNA5 ISA 8.3/8.4.
__device__ __forceinline__ void tdm_load_2d(unsigned lds_off, const void* gaddr,
                                            unsigned tensor_d0, unsigned tensor_d1,
                                            unsigned stride0,
                                            unsigned tile_d0, unsigned tile_d1) {
    unsigned long long ga = (unsigned long long)(size_t)gaddr;
    v4u g0;
    g0[0] = 1u;                                            // count=1 (valid), user mode
    g0[1] = lds_off;                                       // lds_addr (bytes)
    g0[2] = (unsigned)(ga & 0xffffffffu);                  // global_addr[31:0]
    g0[3] = (unsigned)((ga >> 32) & 0x01ffffffu) | (2u << 30);  // addr[56:32] | type=2
    v8i_t g1;
    g1[0] = (int)(1u << 16);                               // data_size=1 (2 bytes)
    g1[1] = (int)((tensor_d0 & 0xffffu) << 16);            // tensor_dim0[15:0]
    g1[2] = (int)(((tensor_d0 >> 16) & 0xffffu) | ((tensor_d1 & 0xffffu) << 16));
    g1[3] = (int)(((tensor_d1 >> 16) & 0xffffu) | ((tile_d0 & 0xffffu) << 16));
    g1[4] = (int)(tile_d1 & 0xffffu);                      // tile_dim1 (tile_dim2=0)
    g1[5] = (int)stride0;                                  // tensor_dim0_stride lo32
    g1[6] = 0;
    g1[7] = 0;
    v4i_t z = {0, 0, 0, 0};                                // groups 2/3 unused (2D)
#if __clang_major__ >= 23
    v8i_t z8 = {0, 0, 0, 0, 0, 0, 0, 0};
    __builtin_amdgcn_tensor_load_to_lds(g0, g1, z, z, z8, 0);
#else
    __builtin_amdgcn_tensor_load_to_lds(g0, g1, z, z, 0);
#endif
}
#endif

// ---------------------------------------------------------------------------
// fp32 -> bf16 elementwise convert
// ---------------------------------------------------------------------------
__global__ void cvt_bf16_kernel(const float* __restrict__ in,
                                unsigned short* __restrict__ out, int n) {
    int i = blockIdx.x * blockDim.x + threadIdx.x;
    int stride = gridDim.x * blockDim.x;
    for (; i < n; i += stride) out[i] = f2bf(in[i]);
}

// ---------------------------------------------------------------------------
// QKV GEMM: y[m,n] = sum_k x[m,k]*w[n,k] + b[n].  One wave computes a 64x32
// output block (4x2 WMMA tiles -> 8 resident v8f accumulators, no spill).
// Scatter into q [B,H,N,HD], k [B,H,N,HD], v TRANSPOSED [B,H,HD,N] (bf16).
// ---------------------------------------------------------------------------
__global__ void __launch_bounds__(256, 1)
qkv_gemm_kernel(const unsigned short* __restrict__ xb,   // [4096,1024]
                const unsigned short* __restrict__ wb,   // [3072,1024]
                const float* __restrict__ bias,          // [3072]
                unsigned short* __restrict__ qb,
                unsigned short* __restrict__ kb,
                unsigned short* __restrict__ vb) {
    const int lane = threadIdx.x & 31;
    const int wave = threadIdx.x >> 5;
    const int tiles_n = QKVO / 32;                    // 96
    int wg = blockIdx.x * (blockDim.x >> 5) + wave;
    int tm = wg / tiles_n, tn = wg % tiles_n;
    if (tm >= ROWS / 64) return;

    const int lo = lane & 15, hi = lane >> 4;
    const int m0 = tm * 64, n0 = tn * 32;

    const unsigned short* arow[4];
    const unsigned short* brow[2];
#pragma unroll
    for (int t = 0; t < 4; ++t)
        arow[t] = xb + (size_t)(m0 + 16 * t + lo) * DIMV;
#pragma unroll
    for (int t = 0; t < 2; ++t)
        brow[t] = wb + (size_t)(n0 + 16 * t + lo) * DIMV;

    v8f c[4][2] = {};
    for (int k0 = 0; k0 < DIMV; k0 += 32) {
        Frag a[4], b[2];
#pragma unroll
        for (int t = 0; t < 4; ++t) {
#pragma unroll
            for (int j = 0; j < 8; ++j) {
                int ka = k0 + 2 * j + 8 * hi + ((j >= 4) ? 8 : 0);
                a[t].s[2 * j]     = arow[t][ka];
                a[t].s[2 * j + 1] = arow[t][ka + 1];
            }
        }
#pragma unroll
        for (int t = 0; t < 2; ++t) {
#pragma unroll
            for (int j = 0; j < 8; ++j) {
                int kn = k0 + 2 * j + 8 * hi + ((j >= 4) ? 8 : 0);
                b[t].s[2 * j]     = brow[t][kn];
                b[t].s[2 * j + 1] = brow[t][kn + 1];
            }
        }
#pragma unroll
        for (int i = 0; i < 4; ++i)
#pragma unroll
            for (int j = 0; j < 2; ++j)
                c[i][j] = WMMA_BF16(a[i].v, b[j].v, c[i][j]);
    }

#pragma unroll
    for (int j = 0; j < 2; ++j) {
        const int n = n0 + 16 * j + lo;
        const float bn = bias[n];
        const int which = n >> 10;          // 0=q 1=k 2=v
        const int rem = n & 1023;
        const int h = rem >> 6, d = rem & 63;
        unsigned short* dstp = (which == 0) ? qb : ((which == 1) ? kb : vb);
#pragma unroll
        for (int i = 0; i < 4; ++i) {
#pragma unroll
            for (int r = 0; r < 8; ++r) {
                int m = m0 + 16 * i + r + 8 * hi;
                int bidx = m >> 11, t = m & 2047;
                size_t dst;
                if (which == 2)   // V transposed: [B,H,HD,SEQ]
                    dst = (((size_t)(bidx * NHEAD + h)) * HDIM + d) * SEQ + t;
                else
                    dst = (((size_t)(bidx * NHEAD + h)) * SEQ + t) * HDIM + d;
                dstp[dst] = f2bf(c[i][j][r] + bn);
            }
        }
    }
}

// ---------------------------------------------------------------------------
// Flash attention.  One block = 4 waves, all on the same (b,h); each wave owns
// one 16-row q tile.  Per 32-key step the K tile (32x64) and Vt tile (64x32)
// are staged into LDS once per block by the Tensor Data Mover (wave 0 issues,
// s_wait_tensorcnt, block barrier), then all waves run WMMA out of LDS.
// ---------------------------------------------------------------------------
__global__ void __launch_bounds__(128, 1)
attention_kernel(const unsigned short* __restrict__ qb,  // [B,H,N,HD]
                 const unsigned short* __restrict__ kb,  // [B,H,N,HD]
                 const unsigned short* __restrict__ vt,  // [B,H,HD,N]
                 unsigned short* __restrict__ wab) {     // [B,N,D]
    __shared__ unsigned short shK[32][64];       // keys x head-dim   (4 KB)
    __shared__ unsigned short shV[64][32];       // head-dim x keys   (4 KB)
    __shared__ unsigned short sh_p[4][16][32];   // per-wave P tile   (4 KB)

    const int lane = threadIdx.x & 31;
    const int wave = threadIdx.x >> 5;
    const int lo = lane & 15, hi = lane >> 4;

    const int bh = blockIdx.x >> 5;              // 32 blocks per (b,h)
    const int grp = blockIdx.x & 31;
    const int q0 = (grp * 4 + wave) * 16;

    const unsigned short* Q  = qb + (size_t)bh * SEQ * HDIM;
    const unsigned short* K  = kb + (size_t)bh * SEQ * HDIM;
    const unsigned short* Vt = vt + (size_t)bh * SEQ * HDIM;   // [64][SEQ]

    // Q A-fragments (K = 0..31 and 32..63 of head dim), loaded once
    Frag aq0, aq1;
    {
        const unsigned short* qr = Q + (size_t)(q0 + lo) * HDIM;
#pragma unroll
        for (int j = 0; j < 8; ++j) {
            int ka = 2 * j + 8 * hi + ((j >= 4) ? 8 : 0);
            aq0.s[2 * j]     = qr[ka];
            aq0.s[2 * j + 1] = qr[ka + 1];
            aq1.s[2 * j]     = qr[32 + ka];
            aq1.s[2 * j + 1] = qr[32 + ka + 1];
        }
    }

    float mrun[8], lrun[8];
    v8f o0 = {}, o1 = {}, o2 = {}, o3 = {};
#pragma unroll
    for (int r = 0; r < 8; ++r) { mrun[r] = -1e30f; lrun[r] = 0.0f; }

    for (int kt = 0; kt < SEQ; kt += 32) {
        __syncthreads();   // previous iteration's LDS reads complete
#if HAVE_TDM
        if (wave == 0) {
            // K rows kt..kt+31, full 64-wide -> shK[32][64]
            tdm_load_2d((unsigned)(size_t)&shK[0][0], K + (size_t)kt * HDIM,
                        /*d0=*/HDIM, /*d1=*/SEQ, /*stride0=*/HDIM,
                        /*tile0=*/HDIM, /*tile1=*/32);
            // Vt rows 0..63, cols kt..kt+31 -> shV[64][32]
            tdm_load_2d((unsigned)(size_t)&shV[0][0], Vt + (size_t)kt,
                        /*d0=*/SEQ, /*d1=*/HDIM, /*stride0=*/SEQ,
                        /*tile0=*/32, /*tile1=*/HDIM);
            __builtin_amdgcn_s_wait_tensorcnt(0);
        }
#else
        for (int idx = threadIdx.x; idx < 32 * 64; idx += 128) {
            int r = idx >> 6, cc = idx & 63;
            shK[r][cc] = K[(size_t)(kt + r) * HDIM + cc];
        }
        for (int idx = threadIdx.x; idx < 64 * 32; idx += 128) {
            int r = idx >> 5, cc = idx & 31;
            shV[r][cc] = Vt[(size_t)r * SEQ + kt + cc];
        }
#endif
        __syncthreads();

        // ---- S = Q K^T for 32 keys (two 16x16 tiles) from shK ----
        v8f s0 = {}, s1 = {};
        {
            Frag b0, b1;
#pragma unroll
            for (int j = 0; j < 8; ++j) {
                int kk = 2 * j + 8 * hi + ((j >= 4) ? 8 : 0);
                b0.s[2 * j]     = shK[lo][kk];
                b0.s[2 * j + 1] = shK[lo][kk + 1];
                b1.s[2 * j]     = shK[lo][32 + kk];
                b1.s[2 * j + 1] = shK[lo][32 + kk + 1];
            }
            s0 = WMMA_BF16(aq0.v, b0.v, s0);
            s0 = WMMA_BF16(aq1.v, b1.v, s0);
#pragma unroll
            for (int j = 0; j < 8; ++j) {
                int kk = 2 * j + 8 * hi + ((j >= 4) ? 8 : 0);
                b0.s[2 * j]     = shK[16 + lo][kk];
                b0.s[2 * j + 1] = shK[16 + lo][kk + 1];
                b1.s[2 * j]     = shK[16 + lo][32 + kk];
                b1.s[2 * j + 1] = shK[16 + lo][32 + kk + 1];
            }
            s1 = WMMA_BF16(aq0.v, b0.v, s1);
            s1 = WMMA_BF16(aq1.v, b1.v, s1);
        }

        // ---- online softmax (row = r + 8*hi, cols across 16-lane half) ----
        float p0[8], p1[8], alpha[8];
#pragma unroll
        for (int r = 0; r < 8; ++r) {
            float a = s0[r] * SCALEF;
            float b = s1[r] * SCALEF;
            float vmax = fmaxf(a, b);
#pragma unroll
            for (int msk = 1; msk < 16; msk <<= 1)
                vmax = fmaxf(vmax, __shfl_xor(vmax, msk, 32));
            float mnew = fmaxf(mrun[r], vmax);
            alpha[r] = __expf(mrun[r] - mnew);
            float e0 = __expf(a - mnew);
            float e1 = __expf(b - mnew);
            p0[r] = e0; p1[r] = e1;
            float rs = e0 + e1;
#pragma unroll
            for (int msk = 1; msk < 16; msk <<= 1)
                rs += __shfl_xor(rs, msk, 32);
            lrun[r] = lrun[r] * alpha[r] + rs;
            mrun[r] = mnew;
        }
#pragma unroll
        for (int r = 0; r < 8; ++r) {
            o0[r] *= alpha[r]; o1[r] *= alpha[r];
            o2[r] *= alpha[r]; o3[r] *= alpha[r];
        }

        // ---- P: C-layout -> LDS -> A-layout (per-wave region, DS in-order) ----
#pragma unroll
        for (int r = 0; r < 8; ++r) {
            sh_p[wave][r + 8 * hi][lo]      = f2bf(p0[r]);
            sh_p[wave][r + 8 * hi][lo + 16] = f2bf(p1[r]);
        }
        Frag ap;
#pragma unroll
        for (int j = 0; j < 8; ++j) {
            int ka = 2 * j + 8 * hi + ((j >= 4) ? 8 : 0);
            ap.s[2 * j]     = sh_p[wave][lo][ka];
            ap.s[2 * j + 1] = sh_p[wave][lo][ka + 1];
        }

        // ---- O += P * V from shV (4 head-dim column groups of 16) ----
#pragma unroll
        for (int g = 0; g < 4; ++g) {
            Frag bv;
#pragma unroll
            for (int j = 0; j < 8; ++j) {
                int kk = 2 * j + 8 * hi + ((j >= 4) ? 8 : 0);
                bv.s[2 * j]     = shV[g * 16 + lo][kk];
                bv.s[2 * j + 1] = shV[g * 16 + lo][kk + 1];
            }
            v8f* op = (g == 0) ? &o0 : ((g == 1) ? &o1 : ((g == 2) ? &o2 : &o3));
            *op = WMMA_BF16(ap.v, bv.v, *op);
        }
    }

    // ---- finalize: wa[b, t, h*64 + d], bf16 ----
    const int bidx = bh >> 4, h = bh & 15;
#pragma unroll
    for (int r = 0; r < 8; ++r) {
        float inv = 1.0f / lrun[r];
        int t = q0 + r + 8 * hi;
        size_t base = ((size_t)bidx * SEQ + t) * DIMV + (size_t)h * HDIM;
        wab[base + 0 * 16 + lo] = f2bf(o0[r] * inv);
        wab[base + 1 * 16 + lo] = f2bf(o1[r] * inv);
        wab[base + 2 * 16 + lo] = f2bf(o2[r] * inv);
        wab[base + 3 * 16 + lo] = f2bf(o3[r] * inv);
    }
}

// ---------------------------------------------------------------------------
// Proj GEMM: out[m,n] = sum_k wa[m,k]*w[n,k] + b[n], fp32 out.  64x32/wave.
// ---------------------------------------------------------------------------
__global__ void __launch_bounds__(256, 1)
proj_gemm_kernel(const unsigned short* __restrict__ wab,  // [4096,1024]
                 const unsigned short* __restrict__ wb,   // [1024,1024]
                 const float* __restrict__ bias,
                 float* __restrict__ out) {
    const int lane = threadIdx.x & 31;
    const int wave = threadIdx.x >> 5;
    const int tiles_n = DIMV / 32;                    // 32
    int wg = blockIdx.x * (blockDim.x >> 5) + wave;
    int tm = wg / tiles_n, tn = wg % tiles_n;
    if (tm >= ROWS / 64) return;

    const int lo = lane & 15, hi = lane >> 4;
    const int m0 = tm * 64, n0 = tn * 32;

    const unsigned short* arow[4];
    const unsigned short* brow[2];
#pragma unroll
    for (int t = 0; t < 4; ++t)
        arow[t] = wab + (size_t)(m0 + 16 * t + lo) * DIMV;
#pragma unroll
    for (int t = 0; t < 2; ++t)
        brow[t] = wb + (size_t)(n0 + 16 * t + lo) * DIMV;

    v8f c[4][2] = {};
    for (int k0 = 0; k0 < DIMV; k0 += 32) {
        Frag a[4], b[2];
#pragma unroll
        for (int t = 0; t < 4; ++t) {
#pragma unroll
            for (int j = 0; j < 8; ++j) {
                int ka = k0 + 2 * j + 8 * hi + ((j >= 4) ? 8 : 0);
                a[t].s[2 * j]     = arow[t][ka];
                a[t].s[2 * j + 1] = arow[t][ka + 1];
            }
        }
#pragma unroll
        for (int t = 0; t < 2; ++t) {
#pragma unroll
            for (int j = 0; j < 8; ++j) {
                int kn = k0 + 2 * j + 8 * hi + ((j >= 4) ? 8 : 0);
                b[t].s[2 * j]     = brow[t][kn];
                b[t].s[2 * j + 1] = brow[t][kn + 1];
            }
        }
#pragma unroll
        for (int i = 0; i < 4; ++i)
#pragma unroll
            for (int j = 0; j < 2; ++j)
                c[i][j] = WMMA_BF16(a[i].v, b[j].v, c[i][j]);
    }

#pragma unroll
    for (int j = 0; j < 2; ++j) {
        const int n = n0 + 16 * j + lo;
        const float bn = bias[n];
#pragma unroll
        for (int i = 0; i < 4; ++i)
#pragma unroll
            for (int r = 0; r < 8; ++r) {
                int m = m0 + 16 * i + r + 8 * hi;
                out[(size_t)m * DIMV + n] = c[i][j][r] + bn;
            }
    }
}

// ---------------------------------------------------------------------------
// Host side
// ---------------------------------------------------------------------------
extern "C" void kernel_launch(void* const* d_in, const int* in_sizes, int n_in,
                              void* d_out, int out_size, void* d_ws, size_t ws_size,
                              hipStream_t stream) {
    const float* x      = (const float*)d_in[0];   // [2,2048,1024]
    const float* qkv_w  = (const float*)d_in[1];   // [3072,1024]
    const float* qkv_b  = (const float*)d_in[2];   // [3072]
    const float* proj_w = (const float*)d_in[3];   // [1024,1024]
    const float* proj_b = (const float*)d_in[4];   // [1024]
    float* out = (float*)d_out;                    // [2,2048,1024]

    // Workspace layout (bf16 buffers), total ~48 MB
    char* w = (char*)d_ws;
    unsigned short* xb    = (unsigned short*)w;   w += (size_t)ROWS * DIMV * 2;
    unsigned short* qkvwb = (unsigned short*)w;   w += (size_t)QKVO * DIMV * 2;
    unsigned short* projwb= (unsigned short*)w;   w += (size_t)DIMV * DIMV * 2;
    unsigned short* qb    = (unsigned short*)w;   w += (size_t)ROWS * DIMV * 2;
    unsigned short* kb    = (unsigned short*)w;   w += (size_t)ROWS * DIMV * 2;
    unsigned short* vtb   = (unsigned short*)w;   w += (size_t)ROWS * DIMV * 2;  // [B,H,HD,SEQ]
    unsigned short* wab   = (unsigned short*)w;   w += (size_t)ROWS * DIMV * 2;
    (void)ws_size; (void)n_in; (void)in_sizes; (void)out_size;

    // 1) convert inputs to bf16
    cvt_bf16_kernel<<<1024, 256, 0, stream>>>(x,      xb,     ROWS * DIMV);
    cvt_bf16_kernel<<<1024, 256, 0, stream>>>(qkv_w,  qkvwb,  QKVO * DIMV);
    cvt_bf16_kernel<<<1024, 256, 0, stream>>>(proj_w, projwb, DIMV * DIMV);

    // 2) QKV GEMM: (4096/64)*(3072/32) = 6144 waves, 8 waves/block
    qkv_gemm_kernel<<<768, 256, 0, stream>>>(xb, qkvwb, qkv_b, qb, kb, vtb);

    // 3) flash attention: 32 (b,h) * 32 blocks, 4 waves/block
    attention_kernel<<<1024, 128, 0, stream>>>(qb, kb, vtb, wab);

    // 4) proj GEMM: (4096/64)*(1024/32) = 2048 waves, 8 waves/block
    proj_gemm_kernel<<<256, 256, 0, stream>>>(wab, projwb, proj_b, out);
}
